// FCSTG_AttnFusion_InceptionNet_56332791054803
// MI455X (gfx1250) — matrix operand order repaired and checked by reading, not compiled, not run on hardware
//
#include <hip/hip_runtime.h>
#include <hip/hip_bf16.h>
#include <cstdint>

// ---------------- problem constants ----------------
#define BS    8
#define T_DIM 64
#define N_DIM 64
#define F_DIM 256
#define TW    8
#define STRIDE 2
#define NW    29            // (64-8)/2+1
#define LQ    512           // TW*N
#define NB    232           // BS*NW
#define OUTC  128
#define EPS_  1e-5f
#define SLOPE 0.01f
#define DECAY 0.7f
#define CNT_F 118784.0f     // NB*LQ

typedef __bf16 bf16_t;
typedef __attribute__((ext_vector_type(16))) __bf16 v16bf;
typedef __attribute__((ext_vector_type(8)))  __bf16 v8bf;
typedef __attribute__((ext_vector_type(8)))  float  v8f;

#define WMMA_BF16(A_, B_, C_) \
  __builtin_amdgcn_wmma_f32_16x16x32_bf16(false, (A_), false, (B_), (short)0, (C_), false, false)

// ---- CDNA5 async global->LDS copy (ASYNCcnt), 16B per lane per issue ------
__device__ __forceinline__ void async_copy_tile(void* ldsDst, const void* gsrc,
                                                int tid, int nu4) {
  uint32_t lbase = (uint32_t)(size_t)ldsDst;       // low 32 bits = LDS byte addr
  const uint4* g = (const uint4*)gsrc;
  for (int i = tid; i < nu4; i += 256) {
    asm volatile("global_load_async_to_lds_b128 %0, %1, off"
                 :: "v"(lbase + i * 16), "v"(g + i)
                 : "memory");
  }
}
__device__ __forceinline__ void wait_async0() {
  asm volatile("s_wait_asynccnt 0x0" ::: "memory");
}

// ---- WMMA fragment loads, all as 2x 16-byte LDS loads ---------------------
__device__ __forceinline__ v16bf load_a_frag(const bf16_t* src, int stride, int m0, int k0) {
  int lane = threadIdx.x & 31;
  int m = m0 + (lane & 15);
  int k = k0 + ((lane >> 4) << 4);
  union { uint4 u[2]; v16bf v; } r;
  r.u[0] = *(const uint4*)(src + m * stride + k);
  r.u[1] = *(const uint4*)(src + m * stride + k + 8);
  return r.v;
}
// B 32x16 bf16 with source stored [n][k]
__device__ __forceinline__ v16bf load_bT_frag(const bf16_t* src, int stride, int k0, int n0) {
  int lane = threadIdx.x & 31;
  int n = n0 + (lane & 15);
  int k = k0 + ((lane >> 4) << 4);
  union { uint4 u[2]; v16bf v; } r;
  r.u[0] = *(const uint4*)(src + n * stride + k);
  r.u[1] = *(const uint4*)(src + n * stride + k + 8);
  return r.v;
}

// window replication weight for timestep t
__device__ __forceinline__ int win_weight(int t) {
  int c = 0;
#pragma unroll
  for (int tw = 0; tw < TW; ++tw) {
    int d = t - tw;
    if (d >= 0 && (d & 1) == 0 && (d >> 1) < NW) ++c;
  }
  return c;
}

// ========== K0: transpose weights to [n][k] bf16 (one-time, tiny) ==========
__global__ void k0_wtrans(const float* __restrict__ Wm, const float* __restrict__ thW,
                          bf16_t* __restrict__ Wt, bf16_t* __restrict__ Tt) {
  for (int i = blockIdx.x * 256 + threadIdx.x; i < F_DIM * F_DIM; i += gridDim.x * 256) {
    int n = i >> 8, k = i & 255;
    Wt[n * F_DIM + k] = (bf16_t)Wm[k * F_DIM + n];
  }
  for (int i = blockIdx.x * 256 + threadIdx.x; i < OUTC * F_DIM; i += gridDim.x * 256) {
    int n = i >> 8, k = i & 255;
    Tt[n * F_DIM + k] = (bf16_t)thW[k * OUTC + n];
  }
}

// ========== K1a/K1b: BN0 stats (deterministic two-stage) ===================
__global__ void k1a_bn0_partial(const float* __restrict__ x,
                                float* __restrict__ p1, float* __restrict__ p2) {
  int bt = blockIdx.x, t = bt & 63, f = threadIdx.x;
  float wt = (float)win_weight(t);
  float s1 = 0.f, s2 = 0.f;
  const float* base = x + (size_t)bt * N_DIM * F_DIM + f;
  for (int n = 0; n < N_DIM; ++n) {
    float v = base[(size_t)n * F_DIM];
    s1 += v; s2 += v * v;
  }
  p1[bt * F_DIM + f] = wt * s1;
  p2[bt * F_DIM + f] = wt * s2;
}

__global__ void k1b_bn0_final(const float* __restrict__ p1, const float* __restrict__ p2,
                              float* __restrict__ mu0, float* __restrict__ inv0) {
  int f = threadIdx.x;
  float s1 = 0.f, s2 = 0.f;
  for (int bt = 0; bt < BS * T_DIM; ++bt) {
    s1 += p1[bt * F_DIM + f];
    s2 += p2[bt * F_DIM + f];
  }
  float m = s1 / CNT_F;
  float v = s2 / CNT_F - m * m;
  mu0[f]  = m;
  inv0[f] = rsqrtf(v + EPS_);
}

// ========== K2: h = x @ W_map + b_map (bf16) ===============================
__global__ void __launch_bounds__(256)
k2_map_gemm(const float* __restrict__ x, const bf16_t* __restrict__ Wt,
            const float* __restrict__ bmap, bf16_t* __restrict__ h_bf) {
  extern __shared__ char smem[];
  bf16_t* xs  = (bf16_t*)smem;                   // [128][256] 64KB
  bf16_t* wts = (bf16_t*)(smem + 65536);         // [256 n][256 k] 128KB
  const int tid = threadIdx.x, lane = tid & 31, wave = tid >> 5;
  const int row0 = blockIdx.x * 128;

  async_copy_tile(wts, Wt, tid, 8192);           // overlaps with convert below
  for (int u = tid; u < 4096; u += 256) {
    int n = u >> 5, f0 = (u & 31) * 8;
    const float4* px = (const float4*)(x + ((size_t)(row0 + n)) * F_DIM + f0);
    float4 a = px[0], b = px[1];
    v8bf pk;
    pk[0] = (bf16_t)a.x; pk[1] = (bf16_t)a.y; pk[2] = (bf16_t)a.z; pk[3] = (bf16_t)a.w;
    pk[4] = (bf16_t)b.x; pk[5] = (bf16_t)b.y; pk[6] = (bf16_t)b.z; pk[7] = (bf16_t)b.w;
    *(v8bf*)(xs + n * F_DIM + f0) = pk;
  }
  wait_async0();
  __syncthreads();

  const int m0 = wave * 16;
  v8f acc[16] = {};
  for (int kt = 0; kt < 8; ++kt) {
    v16bf afr = load_a_frag(xs, F_DIM, m0, kt * 32);
#pragma unroll
    for (int nt = 0; nt < 16; ++nt) {
      v16bf bfr = load_bT_frag(wts, F_DIM, kt * 32, nt * 16);
      acc[nt] = WMMA_BF16(afr, bfr, acc[nt]);
    }
  }
#pragma unroll
  for (int nt = 0; nt < 16; ++nt) {
    int col = nt * 16 + (lane & 15);
    float bb = bmap[col];
#pragma unroll
    for (int r = 0; r < 8; ++r) {
      int m = m0 + r + ((lane >> 4) << 3);
      h_bf[(size_t)(row0 + m) * F_DIM + col] = (bf16_t)(acc[nt][r] + bb);
    }
  }
}

// ========== K3: xn = BN0(x), stored TRANSPOSED per (b,t): [f][n] ===========
__global__ void k3_xnT(const float* __restrict__ x, const float* __restrict__ mu,
                       const float* __restrict__ inv, const float* __restrict__ g,
                       const float* __restrict__ be, bf16_t* __restrict__ xn_t) {
  __shared__ bf16_t Xl[N_DIM * F_DIM];           // [n][f] 32KB
  const int bt = blockIdx.x;
  const int tid = threadIdx.x;
  for (int u = tid; u < 2048; u += 256) {
    int n = u >> 5, f0 = (u & 31) * 8;
    const float4* px = (const float4*)(x + ((size_t)bt * N_DIM + n) * F_DIM + f0);
    float4 a = px[0], b = px[1];
    float vv[8] = { a.x, a.y, a.z, a.w, b.x, b.y, b.z, b.w };
    v8bf pk;
#pragma unroll
    for (int j = 0; j < 8; ++j) {
      int f = f0 + j;
      pk[j] = (bf16_t)((vv[j] - mu[f]) * inv[f] * g[f] + be[f]);
    }
    *(v8bf*)(Xl + n * F_DIM + f0) = pk;
  }
  __syncthreads();
  for (int u = tid; u < 2048; u += 256) {
    int f = u >> 3, n0 = (u & 7) * 8;
    v8bf pk;
#pragma unroll
    for (int j = 0; j < 8; ++j) pk[j] = Xl[(n0 + j) * F_DIM + f];
    *(v8bf*)(xn_t + ((size_t)bt * F_DIM + f) * N_DIM + n0) = pk;
  }
}

// ========== K4: fused attention + A@xn + theta projection ==================
// grid (8 tiles, 232 batches), 256 threads (8 waves), 224KB dynamic LDS,
// async double-buffered chunk staging.
__global__ void __launch_bounds__(256)
k4_attn(const bf16_t* __restrict__ h_bf, const bf16_t* __restrict__ xn_t,
        const bf16_t* __restrict__ Tt, const float* __restrict__ thb,
        float* __restrict__ outw) {
  extern __shared__ char smem[];
  // R1 (96KB @0): phase1 Q(32K)+K0(32K)+K1(32K); phase2 Abf(64K @0); phase3 TtS(64K @0)
  // R2 (128KB @96K): phase1 Ss f32; phase2 Xb0/Xb1 (32K each); AxBf(32K @+64K)
  bf16_t* Qs   = (bf16_t*)smem;
  bf16_t* Ks0  = (bf16_t*)(smem + 32768);
  bf16_t* Ks1  = (bf16_t*)(smem + 65536);
  bf16_t* Abf  = (bf16_t*)smem;                        // [64 m][512 k]
  bf16_t* TtS  = (bf16_t*)smem;                        // [128 n][256 k]
  float*  Ss   = (float*)(smem + 98304);               // [64][512]
  bf16_t* Xb0  = (bf16_t*)(smem + 98304);              // [256 n][64 k]
  bf16_t* Xb1  = (bf16_t*)(smem + 98304 + 32768);
  bf16_t* AxBf = (bf16_t*)(smem + 98304 + 65536);      // [64 m][256 k]

  const int tid = threadIdx.x, lane = tid & 31, wave = tid >> 5;
  const int tile = blockIdx.x;            // query tw (0..7)
  const int q    = blockIdx.y;            // 0..231
  const int b = q / NW, w = q % NW;
  const int tq = STRIDE * w + tile;
  const bf16_t* hb  = h_bf + (size_t)(b * T_DIM) * N_DIM * F_DIM;
  const bf16_t* xnb = xn_t + (size_t)(b * T_DIM) * F_DIM * N_DIM;

  // prologue: async-load Q tile and key chunk 0
  async_copy_tile(Qs,  hb + (size_t)tq * N_DIM * F_DIM, tid, 2048);
  async_copy_tile(Ks0, hb + (size_t)(STRIDE * w) * N_DIM * F_DIM, tid, 2048);
  wait_async0();
  __syncthreads();

  // ---- phase 1: S[64][512] = leaky(QK^T - 1e8*eye), double-buffered K -----
  const int mt1 = wave >> 1;              // wave's row strip (shared by both tiles)
  const int nt1 = (wave & 1) * 2;
  for (int kc = 0; kc < 8; ++kc) {
    if (kc < 7)
      async_copy_tile((kc & 1) ? Ks0 : Ks1,
                      hb + (size_t)(STRIDE * w + kc + 1) * N_DIM * F_DIM, tid, 2048);
    const bf16_t* Kc = (kc & 1) ? Ks1 : Ks0;
    v8f acc0 = {}, acc1 = {};
    for (int kt = 0; kt < 8; ++kt) {
      v16bf afr = load_a_frag(Qs, F_DIM, mt1 * 16, kt * 32);   // shared A-frag
      v16bf b0  = load_bT_frag(Kc, F_DIM, kt * 32, nt1 * 16);
      v16bf b1  = load_bT_frag(Kc, F_DIM, kt * 32, (nt1 + 1) * 16);
      acc0 = WMMA_BF16(afr, b0, acc0);
      acc1 = WMMA_BF16(afr, b1, acc1);
    }
#pragma unroll
    for (int j = 0; j < 2; ++j) {
      const v8f& acc = j ? acc1 : acc0;
      int gcol = kc * 64 + (nt1 + j) * 16 + (lane & 15);
#pragma unroll
      for (int r = 0; r < 8; ++r) {
        int mrow = mt1 * 16 + r + ((lane >> 4) << 3);
        float v = acc[r];
        if (tile * 64 + mrow == gcol) v -= 1e8f;       // -eye*1e8
        v = v > 0.f ? v : SLOPE * v;                   // leaky_relu
        Ss[mrow * LQ + gcol] = v;
      }
    }
    wait_async0();
    __syncthreads();
  }

  // ---- softmax + decay mask + eye -> Abf (bf16, vectorized) ---------------
  {
    int row = tid >> 2, part = tid & 3;   // 4 adjacent lanes per row
    int c0 = part * 128;
    int grow = tile * 64 + row;
    const float4* s4 = (const float4*)(Ss + row * LQ + c0);
    float mx = -3.4e38f;
    for (int i = 0; i < 32; ++i) {
      float4 v = s4[i];
      mx = fmaxf(mx, fmaxf(fmaxf(v.x, v.y), fmaxf(v.z, v.w)));
    }
    mx = fmaxf(mx, __shfl_xor(mx, 1, 32));
    mx = fmaxf(mx, __shfl_xor(mx, 2, 32));
    float sum = 0.f;
    for (int i = 0; i < 32; ++i) {
      float4 v = s4[i];
      sum += __expf(v.x - mx) + __expf(v.y - mx) + __expf(v.z - mx) + __expf(v.w - mx);
    }
    sum += __shfl_xor(sum, 1, 32);
    sum += __shfl_xor(sum, 2, 32);
    float invs = 1.f / sum;
    float dpow[8];
    dpow[0] = 1.f;
#pragma unroll
    for (int d = 1; d < 8; ++d) dpow[d] = dpow[d - 1] * DECAY;
    for (int i = 0; i < 16; ++i) {
      int c = c0 + i * 8;
      int kcc = c >> 6;
      int dd = kcc > tile ? kcc - tile : tile - kcc;
      float mk = invs * dpow[dd];
      v8bf pk;
#pragma unroll
      for (int jj = 0; jj < 8; ++jj) {
        float p = __expf(Ss[row * LQ + c + jj] - mx) * mk;
        if (c + jj == grow) p += 1.f;    // + eye
        pk[jj] = (bf16_t)p;
      }
      *(v8bf*)(Abf + row * LQ + c) = pk;
    }
  }
  __syncthreads();

  // ---- phase 2: ax = A @ xn, double-buffered xn chunks --------------------
  const int mt2 = wave >> 1;
  const int nt2 = (wave & 1) * 8;
  v8f axacc[8] = {};
  async_copy_tile(Xb0, xnb + (size_t)(STRIDE * w) * F_DIM * N_DIM, tid, 2048);
  wait_async0();
  __syncthreads();
  for (int kc = 0; kc < 8; ++kc) {
    if (kc < 7)
      async_copy_tile((kc & 1) ? Xb0 : Xb1,
                      xnb + (size_t)(STRIDE * w + kc + 1) * F_DIM * N_DIM, tid, 2048);
    const bf16_t* Xc = (kc & 1) ? Xb1 : Xb0;
#pragma unroll
    for (int kk = 0; kk < 2; ++kk) {
      v16bf afr = load_a_frag(Abf, LQ, mt2 * 16, kc * 64 + kk * 32);  // shared A-frag
#pragma unroll
      for (int j = 0; j < 8; ++j) {
        v16bf bfr = load_bT_frag(Xc, N_DIM, kk * 32, (nt2 + j) * 16);
        axacc[j] = WMMA_BF16(afr, bfr, axacc[j]);
      }
    }
    wait_async0();
    __syncthreads();
  }

  // ---- phase 3: out = ax @ theta_W + theta_b ------------------------------
  async_copy_tile(TtS, Tt, tid, 4096);            // overlays Abf (reads done)
#pragma unroll
  for (int j = 0; j < 8; ++j) {                   // spill ax to LDS as bf16
    int col = (nt2 + j) * 16 + (lane & 15);
#pragma unroll
    for (int r = 0; r < 8; ++r) {
      int m = mt2 * 16 + r + ((lane >> 4) << 3);
      AxBf[m * F_DIM + col] = (bf16_t)axacc[j][r];
    }
  }
  wait_async0();
  __syncthreads();

  const int mt3 = wave >> 1;
  const int nt3 = (wave & 1) * 4;
  v8f outacc[4] = {};
  for (int kt = 0; kt < 8; ++kt) {
    v16bf afr = load_a_frag(AxBf, F_DIM, mt3 * 16, kt * 32);          // shared A-frag
#pragma unroll
    for (int j = 0; j < 4; ++j) {
      v16bf bfr = load_bT_frag(TtS, F_DIM, kt * 32, (nt3 + j) * 16);
      outacc[j] = WMMA_BF16(afr, bfr, outacc[j]);
    }
  }
  const size_t obase = ((size_t)q * LQ + tile * 64) * OUTC;
#pragma unroll
  for (int j = 0; j < 4; ++j) {
    int col = (nt3 + j) * 16 + (lane & 15);
    float bb = thb[col];
#pragma unroll
    for (int r = 0; r < 8; ++r) {
      int m = mt3 * 16 + r + ((lane >> 4) << 3);
      outw[obase + (size_t)m * OUTC + col] = outacc[j][r] + bb;
    }
  }
}

// ========== K5a/K5b: BN1 stats (deterministic) =============================
__global__ void k5a_bn1_partial(const float* __restrict__ outw,
                                float* __restrict__ p1, float* __restrict__ p2) {
  int q = blockIdx.x, o = threadIdx.x;
  float s1 = 0.f, s2 = 0.f;
  const float* base = outw + (size_t)q * LQ * OUTC + o;
  for (int l = 0; l < LQ; ++l) {
    float v = base[(size_t)l * OUTC];
    s1 += v; s2 += v * v;
  }
  p1[q * OUTC + o] = s1;
  p2[q * OUTC + o] = s2;
}

__global__ void k5b_bn1_final(const float* __restrict__ p1, const float* __restrict__ p2,
                              float* __restrict__ mu1, float* __restrict__ inv1) {
  int o = threadIdx.x;
  float s1 = 0.f, s2 = 0.f;
  for (int q = 0; q < NB; ++q) {
    s1 += p1[q * OUTC + o];
    s2 += p2[q * OUTC + o];
  }
  float m = s1 / CNT_F;
  float v = s2 / CNT_F - m * m;
  mu1[o]  = m;
  inv1[o] = rsqrtf(v + EPS_);
}

// ========== K7: BN1 + leaky + mean-pool over tw ============================
__global__ void k7_pool(const float* __restrict__ outw, const float* __restrict__ mu1,
                        const float* __restrict__ inv1, const float* __restrict__ g1,
                        const float* __restrict__ b1, float* __restrict__ out) {
  int qn = blockIdx.x;
  int o  = threadIdx.x;
  int q = qn >> 6, n = qn & 63;
  float m = mu1[o], iv = inv1[o], gg = g1[o], bb = b1[o];
  float acc = 0.f;
#pragma unroll
  for (int tw = 0; tw < TW; ++tw) {
    float v = outw[((size_t)q * LQ + tw * 64 + n) * OUTC + o];
    v = (v - m) * iv * gg + bb;
    acc += (v > 0.f ? v : SLOPE * v);
  }
  out[(size_t)qn * OUTC + o] = acc * (1.f / TW);
}

// =============================== launch ====================================
extern "C" void kernel_launch(void* const* d_in, const int* in_sizes, int n_in,
                              void* d_out, int out_size, void* d_ws, size_t ws_size,
                              hipStream_t stream) {
  (void)in_sizes; (void)n_in; (void)out_size; (void)ws_size;
  const float* x    = (const float*)d_in[0];
  const float* Wm   = (const float*)d_in[1];
  const float* bmap = (const float*)d_in[2];
  const float* g0   = (const float*)d_in[3];
  const float* be0  = (const float*)d_in[4];
  const float* thW  = (const float*)d_in[5];
  const float* thb  = (const float*)d_in[6];
  const float* g1   = (const float*)d_in[7];
  const float* be1  = (const float*)d_in[8];
  float* out = (float*)d_out;

  // workspace layout
  float* p1   = (float*)d_ws;                    // 512*256
  float* p2   = p1 + 512 * F_DIM;
  float* p1b  = p2 + 512 * F_DIM;                // 232*128
  float* p2b  = p1b + NB * OUTC;
  float* mu0  = p2b + NB * OUTC;                 // 256
  float* inv0 = mu0 + F_DIM;
  float* mu1  = inv0 + F_DIM;                    // 128
  float* inv1 = mu1 + OUTC;
  bf16_t* Wt   = (bf16_t*)(inv1 + OUTC);                         // [256 n][256 k]
  bf16_t* Tt   = Wt + (size_t)F_DIM * F_DIM;                     // [128 n][256 k]
  bf16_t* h_bf = Tt + (size_t)OUTC * F_DIM;                      // 8*64*64*256
  bf16_t* xn_t = h_bf + (size_t)BS * T_DIM * N_DIM * F_DIM;      // [b][t][f][n]
  float*  outw = (float*)(xn_t + (size_t)BS * T_DIM * N_DIM * F_DIM); // 232*512*128

  hipFuncSetAttribute((const void*)k2_map_gemm,
                      hipFuncAttributeMaxDynamicSharedMemorySize, 196608);
  hipFuncSetAttribute((const void*)k4_attn,
                      hipFuncAttributeMaxDynamicSharedMemorySize, 229376);

  k0_wtrans<<<64, 256, 0, stream>>>(Wm, thW, Wt, Tt);
  k1a_bn0_partial<<<512, 256, 0, stream>>>(x, p1, p2);
  k1b_bn0_final<<<1, 256, 0, stream>>>(p1, p2, mu0, inv0);
  k2_map_gemm<<<256, 256, 196608, stream>>>(x, Wt, bmap, h_bf);
  k3_xnT<<<512, 256, 0, stream>>>(x, mu0, inv0, g0, be0, xn_t);
  k4_attn<<<dim3(8, NB), 256, 229376, stream>>>(h_bf, xn_t, Tt, thb, outw);
  k5a_bn1_partial<<<NB, 128, 0, stream>>>(outw, p1b, p2b);
  k5b_bn1_final<<<1, 128, 0, stream>>>(p1b, p2b, mu1, inv1);
  k7_pool<<<NB * N_DIM, 128, 0, stream>>>(outw, mu1, inv1, g1, be1, out);
}
